// DVGMatchModule_66477503808125
// MI455X (gfx1250) — compile-verified, matrix operand order checked by reading, not run on hardware
//
#include <hip/hip_runtime.h>
#include <hip/hip_bf16.h>

// ---------------------------------------------------------------------------
// CDNA5 (gfx1250) implementation of the DVG match module.
// All GEMMs (QKV/O projections, Q@K^T, P@V, match-head linears) run on
// v_wmma_f32_16x16x32_f16. f16 operands that live in LDS are stored
// pre-swizzled into WMMA fragment order so each lane's 16 halves are one
// contiguous 32-byte chunk (2x ds_load_b128 per fragment).
// ---------------------------------------------------------------------------

typedef __attribute__((ext_vector_type(16))) _Float16 v16h;
typedef __attribute__((ext_vector_type(8)))  float    v8f;

#define Bv 16
#define Kv 256
#define Lv 16
#define Tv 32
#define Dv 128
#define Hv 4

// K-index inside a 16-half WMMA fragment for element i, lane-half h.
// ISA 7.12.2 16-bit A 16x32: lanes 0-15: VGPR0..3 -> K0..7, VGPR4..7 -> K16..23
//                            lanes 16-31: K8..15 / K24..31
__device__ __forceinline__ int kmap16(int i, int h) {
  return (i < 8) ? (8 * h + i) : (16 + 8 * h + (i - 8));
}

// Load 16 halves from a row-contiguous f32 row: elements p[8h..8h+7] and
// p[16+8h..16+8h+7] (used for A fragments and K^T B-fragments).
__device__ __forceinline__ v16h load_frag_rows_f32(const float* __restrict__ p, int hh) {
  float4 x0 = *(const float4*)(p + 8 * hh);
  float4 x1 = *(const float4*)(p + 8 * hh + 4);
  float4 y0 = *(const float4*)(p + 8 * hh + 16);
  float4 y1 = *(const float4*)(p + 8 * hh + 20);
  v16h r;
  r[0] = (_Float16)x0.x;  r[1] = (_Float16)x0.y;
  r[2] = (_Float16)x0.z;  r[3] = (_Float16)x0.w;
  r[4] = (_Float16)x1.x;  r[5] = (_Float16)x1.y;
  r[6] = (_Float16)x1.z;  r[7] = (_Float16)x1.w;
  r[8] = (_Float16)y0.x;  r[9] = (_Float16)y0.y;
  r[10] = (_Float16)y0.z; r[11] = (_Float16)y0.w;
  r[12] = (_Float16)y1.x; r[13] = (_Float16)y1.y;
  r[14] = (_Float16)y1.z; r[15] = (_Float16)y1.w;
  return r;
}

// A-fragment (16x32) from f32 row-major memory. base -> (row0, k0).
__device__ __forceinline__ v16h load_a_f32(const float* __restrict__ base, int ld) {
  int lane = threadIdx.x & 31;
  return load_frag_rows_f32(base + (size_t)(lane & 15) * ld, lane >> 4);
}

// B^T fragment: need B[k][n] where memory is [n][k] (keys are rows).
__device__ __forceinline__ v16h load_bT_f32(const float* __restrict__ base, int ld) {
  int lane = threadIdx.x & 31;
  return load_frag_rows_f32(base + (size_t)(lane & 15) * ld, lane >> 4);
}

// Gather a B-fragment lane (16 halves) from f32 memory laid out [k][n];
// base pre-offset to (k0, n). Used only for one-time LDS staging.
__device__ __forceinline__ v16h gather_b_f32(const float* __restrict__ base, int ld, int hh) {
  v16h t;
#pragma unroll
  for (int i = 0; i < 16; ++i) t[i] = (_Float16)base[(size_t)kmap16(i, hh) * ld];
  return t;
}

__device__ __forceinline__ v8f wmma_f16(v16h a, v16h b, v8f c) {
  return __builtin_amdgcn_wmma_f32_16x16x32_f16(false, a, false, b, (short)0, c,
                                                false, false);
}

// ---------------------------------------------------------------------------
// Distance bias: one block per (b, j) column of the K x K matrix.
// bias layout: (B, 2, K, K); heads 2,3 are implicit zeros (skipped in attn).
// ---------------------------------------------------------------------------
__global__ void bias_kernel(const float* __restrict__ center, float* __restrict__ bias) {
  __shared__ float sred[256];
  int b = blockIdx.x >> 8;
  int j = blockIdx.x & 255;
  int i = threadIdx.x;
  float cx = center[(b * 256 + j) * 3 + 0] - center[(b * 256 + i) * 3 + 0];
  float cy = center[(b * 256 + j) * 3 + 1] - center[(b * 256 + i) * 3 + 1];
  float cz = center[(b * 256 + j) * 3 + 2] - center[(b * 256 + i) * 3 + 2];
  float dist = sqrtf(cx * cx + cy * cy + cz * cz);
  float dw = 1.0f / (dist + 0.01f);
  sred[i] = dw;
  __syncthreads();
  for (int o = 128; o > 0; o >>= 1) {
    if (i < o) sred[i] += sred[i + o];
    __syncthreads();
  }
  float inv = 1.0f / sred[0];
  bias[(((size_t)b * 2 + 0) * 256 + i) * 256 + j] = dw * inv;
  bias[(((size_t)b * 2 + 1) * 256 + i) * 256 + j] = -dist;
}

// ---------------------------------------------------------------------------
// Y[M,128] = X[M,128] @ W[128,128] + bvec.
// W staged f16 in LDS in WMMA fragment order: [ks(4)][ct(8)][lane(32)][i(16)].
// 256 threads = 8 waves; block tile 64 rows x 128 cols; 4 WMMA tiles/wave.
// ---------------------------------------------------------------------------
__global__ void linear_kernel(const float* __restrict__ X, const float* __restrict__ W,
                              const float* __restrict__ bvec, float* __restrict__ Y,
                              int M) {
  __shared__ _Float16 Wl[128 * 128];  // 32 KB, fragment-swizzled
  int tid = threadIdx.x;
  for (int g = tid; g < 1024; g += 256) {
    int ks = g >> 8, ct = (g >> 5) & 7, l = g & 31;
    v16h t = gather_b_f32(W + (size_t)(ks * 32) * 128 + ct * 16 + (l & 15), 128, l >> 4);
    *(v16h*)&Wl[g * 16] = t;
  }
  __syncthreads();

  int wave = tid >> 5, lane = tid & 31;
  int row0 = blockIdx.x * 64 + (wave >> 1) * 16;

  v8f acc[4];
#pragma unroll
  for (int j = 0; j < 4; ++j) acc[j] = (v8f){};

#pragma unroll
  for (int ks = 0; ks < 4; ++ks) {
    v16h a = load_a_f32(X + (size_t)row0 * 128 + ks * 32, 128);
#pragma unroll
    for (int j = 0; j < 4; ++j) {
      int ct = (wave & 1) * 4 + j;
      v16h b = *(const v16h*)&Wl[((ks * 8 + ct) * 32 + lane) * 16];
      acc[j] = wmma_f16(a, b, acc[j]);
    }
  }

  int n = lane & 15, hm = (lane >> 4) * 8;
#pragma unroll
  for (int j = 0; j < 4; ++j) {
    int c = (wave & 1) * 64 + 16 * j + n;
    float bb = bvec[c];
#pragma unroll
    for (int r = 0; r < 8; ++r)
      Y[(size_t)(row0 + hm + r) * 128 + c] = acc[j][r] + bb;
  }
}

// ---------------------------------------------------------------------------
// Attention core: one block per (bn, head, 64-query chunk).
// scores (WMMA, single K-step since DK=32) -> LDS softmax -> P@V (WMMA).
// P (probabilities) and V are kept in LDS in WMMA fragment order so the
// P@V phase issues contiguous ds_load_b128s.
// Dynamic LDS: S f32[64*nk] | P f16[64*nk] (swizzled) | Vl f16[nk*32] (swizzled)
// ---------------------------------------------------------------------------
__global__ void attn_kernel(const float* __restrict__ Q, const float* __restrict__ Km,
                            const float* __restrict__ V, float* __restrict__ Out,
                            const float* __restrict__ bias, int bias_div,
                            const unsigned char* __restrict__ mask, int nk) {
  extern __shared__ unsigned char smem[];
  float* S = (float*)smem;                                        // [64][nk]
  _Float16* P = (_Float16*)(smem + (size_t)64 * nk * 4);          // [4][ksteps][32][16]
  _Float16* Vl = (_Float16*)(smem + (size_t)64 * nk * 6);         // [ksteps][2][32][16]
  __shared__ float rbuf[512];

  int tid = threadIdx.x, wave = tid >> 5, lane = tid & 31;
  int qt = blockIdx.x & 3;
  int h = (blockIdx.x >> 2) & 3;
  int bn = blockIdx.x >> 4;
  int q0 = qt * 64;
  int nkt = nk >> 4;
  int ksteps = nk >> 5;
  int n = lane & 15, hm = (lane >> 4) * 8;
  const float scale = 0.17677669529663687f;  // 1/sqrt(32)

  // ---- stage V fragments into LDS (fragment-swizzled, shared by all waves) ----
  for (int g = tid; g < ksteps * 64; g += 256) {
    int ks = g >> 6, dt = (g >> 5) & 1, l = g & 31;
    const float* vb = V + ((size_t)(bn * nk + ks * 32)) * 128 + h * 32 + dt * 16 + (l & 15);
    v16h t = gather_b_f32(vb, 128, l >> 4);
    *(v16h*)&Vl[g * 16] = t;
  }

  // ---- scores ----
  for (int t = wave; t < 4 * nkt; t += 8) {
    int q2 = t / nkt, kt = t % nkt;
    const float* qbase = Q + ((size_t)(bn * 256 + q0 + q2 * 16)) * 128 + h * 32;
    const float* kbase = Km + ((size_t)(bn * nk + kt * 16)) * 128 + h * 32;
    v16h a = load_a_f32(qbase, 128);
    v16h b = load_bT_f32(kbase, 128);
    v8f c = (v8f){};
    c = wmma_f16(a, b, c);
    int kcol = kt * 16 + n;
#pragma unroll
    for (int r = 0; r < 8; ++r) {
      int qrow = q0 + q2 * 16 + hm + r;  // global query index (0..255)
      float s = c[r] * scale;
      if (bias != nullptr && h < 2) {
        int b2 = bn / bias_div;
        s += bias[(((size_t)b2 * 2 + h) * 256 + qrow) * 256 + kcol];
      }
      if (mask != nullptr && mask[(size_t)bn * nk + kcol]) s = -1e9f;
      S[(q2 * 16 + hm + r) * nk + kcol] = s;
    }
  }
  __syncthreads();

  // ---- softmax per row (64 rows, 4 threads/row); write P in fragment order ----
  {
    int row = tid >> 2, q4 = tid & 3;
    int cw = nk >> 2;
    float* sr = S + row * nk + q4 * cw;
    float mx = -3.4e38f;
    for (int j = 0; j < cw; ++j) mx = fmaxf(mx, sr[j]);
    rbuf[tid] = mx;
    __syncthreads();
    mx = fmaxf(fmaxf(rbuf[row * 4 + 0], rbuf[row * 4 + 1]),
               fmaxf(rbuf[row * 4 + 2], rbuf[row * 4 + 3]));
    float sum = 0.f;
    for (int j = 0; j < cw; ++j) {
      float e = __expf(sr[j] - mx);
      sr[j] = e;
      sum += e;
    }
    rbuf[256 + tid] = sum;
    __syncthreads();
    sum = rbuf[256 + row * 4 + 0] + rbuf[256 + row * 4 + 1] +
          rbuf[256 + row * 4 + 2] + rbuf[256 + row * 4 + 3];
    float inv = 1.0f / sum;
    int q2 = row >> 4, m = row & 15;
    for (int j = 0; j < cw; ++j) {
      int col = q4 * cw + j;
      int ks = col >> 5, kl = col & 31;
      int hh = (kl >> 3) & 1;
      int ii = (kl & 7) + 8 * (kl >> 4);  // inverse of kmap16
      P[(((q2 * ksteps + ks) * 32) + m + 16 * hh) * 16 + ii] = (_Float16)(sr[j] * inv);
    }
  }
  __syncthreads();

  // ---- O = P @ V : 64 x 32 (4 q-tiles x 2 d-tiles = 8 waves) ----
  {
    int q2 = wave >> 1, dt = wave & 1;
    v8f c = (v8f){};
    for (int ks = 0; ks < ksteps; ++ks) {
      v16h a = *(const v16h*)&P[((q2 * ksteps + ks) * 32 + lane) * 16];
      v16h b = *(const v16h*)&Vl[((ks * 2 + dt) * 32 + lane) * 16];
      c = wmma_f16(a, b, c);
    }
    int col = h * 32 + dt * 16 + n;
#pragma unroll
    for (int r = 0; r < 8; ++r)
      Out[((size_t)(bn * 256 + q0 + q2 * 16 + hm + r)) * 128 + col] = c[r];
  }
}

// ---------------------------------------------------------------------------
// out = LayerNorm(xres + yproj) * g + b ; one wave per 128-wide row.
// ---------------------------------------------------------------------------
__global__ void ln_residual_kernel(const float* __restrict__ Xres,
                                   const float* __restrict__ Yp,
                                   const float* __restrict__ g,
                                   const float* __restrict__ b,
                                   float* __restrict__ Out, int M) {
  int wave = threadIdx.x >> 5, lane = threadIdx.x & 31;
  int row = blockIdx.x * 8 + wave;
  if (row >= M) return;
  const float* xr = Xres + (size_t)row * 128;
  const float* yp = Yp + (size_t)row * 128;
  float v[4], s = 0.f, s2 = 0.f;
#pragma unroll
  for (int j = 0; j < 4; ++j) {
    int c = lane + 32 * j;
    v[j] = xr[c] + yp[c];
    s += v[j];
    s2 += v[j] * v[j];
  }
#pragma unroll
  for (int o = 16; o > 0; o >>= 1) {
    s += __shfl_xor(s, o, 32);
    s2 += __shfl_xor(s2, o, 32);
  }
  float mean = s * (1.0f / 128.0f);
  float var = s2 * (1.0f / 128.0f) - mean * mean;
  float inv = rsqrtf(var + 1e-5f);
  float* out = Out + (size_t)row * 128;
#pragma unroll
  for (int j = 0; j < 4; ++j) {
    int c = lane + 32 * j;
    out[c] = (v[j] - mean) * inv * g[c] + b[c];
  }
}

// f (B,K,128) -> f1 (B*L,K,128), broadcast along L.
__global__ void tile_kernel(const float* __restrict__ f, float* __restrict__ f1,
                            int total) {
  const int chunk = Kv * Dv;
  for (int idx = blockIdx.x * blockDim.x + threadIdx.x; idx < total;
       idx += gridDim.x * blockDim.x) {
    int bn = idx / chunk, rem = idx - bn * chunk;
    int b = bn >> 4;  // bn / L
    f1[idx] = f[(size_t)b * chunk + rem];
  }
}

// BatchNorm(eval, mean=0, var=1) + PReLU, in place.
__global__ void bn_prelu_kernel(float* __restrict__ X, const float* __restrict__ g,
                                const float* __restrict__ b,
                                const float* __restrict__ a, int total) {
  float alpha = a[0];
  const float sc = 0.9999950000374997f;  // 1/sqrt(1+1e-5)
  for (int idx = blockIdx.x * blockDim.x + threadIdx.x; idx < total;
       idx += gridDim.x * blockDim.x) {
    int c = idx & 127;
    float y = X[idx] * sc * g[c] + b[c];
    X[idx] = (y >= 0.f) ? y : alpha * y;
  }
}

// confidence[row] = X[row,:] . w + b0 ; one wave per row.
__global__ void head_out_kernel(const float* __restrict__ X, const float* __restrict__ w,
                                const float* __restrict__ b0, float* __restrict__ out,
                                int M) {
  int wave = threadIdx.x >> 5, lane = threadIdx.x & 31;
  int row = blockIdx.x * 8 + wave;
  if (row >= M) return;
  float s = 0.f;
#pragma unroll
  for (int j = 0; j < 4; ++j) {
    int c = lane + 32 * j;
    s += X[(size_t)row * 128 + c] * w[c];
  }
#pragma unroll
  for (int o = 16; o > 0; o >>= 1) s += __shfl_xor(s, o, 32);
  if (lane == 0) out[row] = s + b0[0];
}

// ---------------------------------------------------------------------------
extern "C" void kernel_launch(void* const* d_in, const int* in_sizes, int n_in,
                              void* d_out, int out_size, void* d_ws, size_t ws_size,
                              hipStream_t stream) {
  (void)in_sizes; (void)n_in; (void)out_size; (void)ws_size;

  const float* center   = (const float*)d_in[0];
  const float* features = (const float*)d_in[1];
  const float* lang     = (const float*)d_in[2];
  const unsigned char* amask = (const unsigned char*)d_in[3];
  const float* Wq = (const float*)d_in[4];
  const float* bq = (const float*)d_in[5];
  const float* Wk = (const float*)d_in[6];
  const float* bk = (const float*)d_in[7];
  const float* Wv = (const float*)d_in[8];
  const float* bv = (const float*)d_in[9];
  const float* Wo = (const float*)d_in[10];
  const float* bo = (const float*)d_in[11];
  const float* lng = (const float*)d_in[12];
  const float* lnb = (const float*)d_in[13];
  const float* mW1 = (const float*)d_in[14];
  const float* mb1 = (const float*)d_in[15];
  const float* bn1g = (const float*)d_in[16];
  const float* bn1b = (const float*)d_in[17];
  const float* pr1  = (const float*)d_in[18];
  const float* mW2 = (const float*)d_in[19];
  const float* mb2 = (const float*)d_in[20];
  const float* bn2g = (const float*)d_in[21];
  const float* bn2b = (const float*)d_in[22];
  const float* pr2  = (const float*)d_in[23];
  const float* mW3 = (const float*)d_in[24];
  const float* mb3 = (const float*)d_in[25];

  const int M0 = Bv * Kv;        // 4096
  const int ML = Bv * Lv * Tv;   // 8192
  const int M1 = Bv * Lv * Kv;   // 65536
  const size_t DD = (size_t)Dv * Dv;

  float* ws = (float*)d_ws;
  size_t off = 0;
  auto alloc = [&](size_t nf) { float* p = ws + off; off += nf; return p; };
  float* bias = alloc((size_t)Bv * 2 * Kv * Kv);
  float* bQ = alloc((size_t)M1 * Dv);
  float* bK = alloc((size_t)M1 * Dv);
  float* bV = alloc((size_t)M1 * Dv);
  float* bA = alloc((size_t)M1 * Dv);
  float* bP = alloc((size_t)M1 * Dv);
  float* f  = alloc((size_t)M0 * Dv);
  float* f1 = alloc((size_t)M1 * Dv);

  auto lin = [&](const float* X, const float* W, const float* bvec, float* Y, int M) {
    linear_kernel<<<M / 64, 256, 0, stream>>>(X, W, bvec, Y, M);
  };
  const size_t sh_self  = (size_t)448 * 256;  // S f32 + P f16 + Vl f16 = 112 KB
  const size_t sh_cross = (size_t)448 * 32;   // 14 KB

  // distance bias
  bias_kernel<<<Bv * Kv, 256, 0, stream>>>(center, bias);

  // layer 0: self-attention on scene features (Bn = 16)
  lin(features, Wq + 0 * DD, bq + 0 * Dv, bQ, M0);
  lin(features, Wk + 0 * DD, bk + 0 * Dv, bK, M0);
  lin(features, Wv + 0 * DD, bv + 0 * Dv, bV, M0);
  attn_kernel<<<Bv * Hv * 4, 256, sh_self, stream>>>(bQ, bK, bV, bA, bias, 1, nullptr, 256);
  lin(bA, Wo + 0 * DD, bo + 0 * Dv, bP, M0);
  ln_residual_kernel<<<M0 / 8, 256, 0, stream>>>(features, bP, lng + 0 * Dv, lnb + 0 * Dv, f, M0);

  // tile per referring expression
  tile_kernel<<<2048, 256, 0, stream>>>(f, f1, M1 * Dv);

  // layer 1: cross-attention vs language (Bn = 256, nk = 32)
  lin(f1, Wq + 1 * DD, bq + 1 * Dv, bQ, M1);
  lin(lang, Wk + 1 * DD, bk + 1 * Dv, bK, ML);
  lin(lang, Wv + 1 * DD, bv + 1 * Dv, bV, ML);
  attn_kernel<<<Bv * Lv * Hv * 4, 256, sh_cross, stream>>>(bQ, bK, bV, bA, nullptr, 1, amask, 32);
  lin(bA, Wo + 1 * DD, bo + 1 * Dv, bP, M1);
  ln_residual_kernel<<<M1 / 8, 256, 0, stream>>>(f1, bP, lng + 1 * Dv, lnb + 1 * Dv, f1, M1);

  // layer 2: self-attention with tiled bias (b = bn / L)
  lin(f1, Wq + 2 * DD, bq + 2 * Dv, bQ, M1);
  lin(f1, Wk + 2 * DD, bk + 2 * Dv, bK, M1);
  lin(f1, Wv + 2 * DD, bv + 2 * Dv, bV, M1);
  attn_kernel<<<Bv * Lv * Hv * 4, 256, sh_self, stream>>>(bQ, bK, bV, bA, bias, Lv, nullptr, 256);
  lin(bA, Wo + 2 * DD, bo + 2 * Dv, bP, M1);
  ln_residual_kernel<<<M1 / 8, 256, 0, stream>>>(f1, bP, lng + 2 * Dv, lnb + 2 * Dv, f1, M1);

  // layer 3: cross-attention vs language
  lin(f1, Wq + 3 * DD, bq + 3 * Dv, bQ, M1);
  lin(lang, Wk + 3 * DD, bk + 3 * Dv, bK, ML);
  lin(lang, Wv + 3 * DD, bv + 3 * Dv, bV, ML);
  attn_kernel<<<Bv * Lv * Hv * 4, 256, sh_cross, stream>>>(bQ, bK, bV, bA, nullptr, 1, amask, 32);
  lin(bA, Wo + 3 * DD, bo + 3 * Dv, bP, M1);
  ln_residual_kernel<<<M1 / 8, 256, 0, stream>>>(f1, bP, lng + 3 * Dv, lnb + 3 * Dv, f1, M1);

  // match head: Linear->BN->PReLU x2, then 128->1
  lin(f1, mW1, mb1, bA, M1);
  bn_prelu_kernel<<<2048, 256, 0, stream>>>(bA, bn1g, bn1b, pr1, M1 * Dv);
  lin(bA, mW2, mb2, bP, M1);
  bn_prelu_kernel<<<2048, 256, 0, stream>>>(bP, bn2g, bn2b, pr2, M1 * Dv);
  head_out_kernel<<<M1 / 8, 256, 0, stream>>>(bP, mW3, mb3, (float*)d_out, M1);
}
